// WSOVODRPN_V2_68083821576582
// MI455X (gfx1250) — matrix operand order repaired and checked by reading, not compiled, not run on hardware
//
#include <hip/hip_runtime.h>
#include <hip/hip_bf16.h>
#include <math.h>

// ---------------------------------------------------------------------------
// Types for CDNA5 WMMA
// ---------------------------------------------------------------------------
typedef __attribute__((ext_vector_type(16))) __bf16 v16bf;
typedef __attribute__((ext_vector_type(8)))  float  v8f;

#define NBATCH   2
#define CIN      256
#define KTOT     2304      // 256 * 9
#define KSTEPS   72        // 2304 / 32
#define NLEV     5
#define A_ANCH   3
#define MTOT     4741      // 1000+1000+1000+1000+741
#define POSTK    1000
#define NEG_INF  (-1e9f)
#define SCALE_CLAMP 4.135166556742356f   // log(1000/16)
#define IMG_W    1216.0f
#define IMG_H    800.0f
#define NPIX     64        // pixel tile per block
#define NTHR     256       // 8 wave32

__device__ __forceinline__ unsigned short f2bf(float f) {
  unsigned int u = __float_as_uint(f);
  u += 0x7FFFu + ((u >> 16) & 1u);     // round-to-nearest-even
  return (unsigned short)(u >> 16);
}
__device__ __forceinline__ unsigned int pk2bf(float lo, float hi) {
  return (unsigned int)f2bf(lo) | ((unsigned int)f2bf(hi) << 16);
}
__device__ __forceinline__ unsigned int fkey(float s) {
  unsigned int u = __float_as_uint(s);
  return u ^ ((u >> 31) ? 0xFFFFFFFFu : 0x80000000u);  // order-preserving
}

// ---------------------------------------------------------------------------
// Kernel 1a: pack conv weights into bf16 A-fragments in the CDNA5 16-bit 16x32
// A-matrix lane layout. K ordering: k = tap*256 + c  (tap outer, channel inner)
// Fragment (ks, mt): 32 lanes x 16 halves, contiguous per lane.
// ---------------------------------------------------------------------------
__global__ void pack_wA_kernel(const float* __restrict__ conv_w,
                               unsigned short* __restrict__ wA) {
  int e = blockIdx.x * 256 + threadIdx.x;     // one half per thread
  if (e >= KSTEPS * 16 * 32 * 16) return;
  int h    = e & 15;
  int lane = (e >> 4) & 31;
  int frag = e >> 9;                          // ks*16 + mt
  int ks   = frag >> 4;
  int mt   = frag & 15;
  int m    = mt * 16 + (lane & 15);
  int kk;
  if (lane < 16) kk = (h < 8) ? h       : (h + 8);    // K=0..7  / K=16..23
  else           kk = (h < 8) ? (h + 8) : (h + 16);   // K=8..15 / K=24..31
  int k   = ks * 32 + kk;
  int tap = k >> 8;                           // 0..8
  int c   = k & 255;
  wA[e] = f2bf(conv_w[m * KTOT + c * 9 + tap]);
}

// ---------------------------------------------------------------------------
// Kernel 1b: pack 1x1 head weights (3 obj + 12 delta rows, padded to 16) into
// bf16 A-fragments, K = channel (8 chunks of 32).  8 frags x 512 halves = 8KB.
// ---------------------------------------------------------------------------
__global__ void pack_headA_kernel(const float* __restrict__ obj_w,
                                  const float* __restrict__ delta_w,
                                  unsigned short* __restrict__ hA) {
  int e = blockIdx.x * 256 + threadIdx.x;     // 8*512 = 4096 halves
  if (e >= 8 * 512) return;
  int h    = e & 15;
  int lane = (e >> 4) & 31;
  int ks2  = e >> 9;
  int m    = lane & 15;
  int kk;
  if (lane < 16) kk = h;
  else           kk = h + 16;
  int c = ks2 * 32 + kk;
  float v = 0.f;
  if (m < 3)       v = obj_w[m * CIN + c];
  else if (m < 15) v = delta_w[(m - 3) * CIN + c];
  hA[e] = f2bf(v);
}

// ---------------------------------------------------------------------------
// Kernel 2: fused implicit-GEMM 3x3 conv (256->256) + ReLU + WMMA 1x1 heads.
// Block = 256 threads (8 wave32). Block computes 64 pixels x 256 out-chan.
// Wave w: M-tiles {2w,2w+1} x N-tiles {0..3} -> 8 WMMAs per K-step per wave.
// Staging is software-pipelined: chunk cc+1's 8 gather loads issue while
// chunk cc's WMMAs run. ldsB double-buffered -> one barrier per K-step.
// Heads: t scattered to LDS in B-fragment layout; waves 0..3 run an 8-step
// WMMA loop (15x256 x 256x16 each) and store sc/dl from C registers.
// ---------------------------------------------------------------------------
__global__ __launch_bounds__(NTHR)
void conv_rpn_wmma_kernel(const float* __restrict__ feat,
                          const unsigned short* __restrict__ wA,
                          const unsigned short* __restrict__ headA,
                          const float* __restrict__ conv_b,
                          const float* __restrict__ obj_b,
                          const float* __restrict__ delta_b,
                          float* __restrict__ sc,   // [N][HW*A]
                          float* __restrict__ dl,   // [N][HW*A][4]
                          int H, int W) {
  __shared__ __align__(32) unsigned short ldsB[2 * 2048];   // 2 x (32K x 64pix)
  __shared__ __align__(32) unsigned short ldsT[8 * 4 * 512]; // t, frag layout

  const int tid  = threadIdx.x;
  const int wave = tid >> 5;
  const int lane = tid & 31;
  const int HW    = H * W;
  const int total = NBATCH * HW;
  const int P0    = blockIdx.x * NPIX;

  // ---- per-thread staging geometry, fixed for the whole K loop ----
  const int pix   = tid & 63;            // this thread's staged pixel
  const int klb   = tid >> 6;            // 0..3 ; thread stages kl = 8*klb+j
  const int p     = P0 + pix;
  const bool pval = p < total;
  const int pc    = pval ? p : (total - 1);
  const int n     = pc / HW;
  const int hw    = pc - n * HW;
  const int x     = hw % W;
  const int y     = hw / W;
  const int g     = pix >> 4;            // N-tile of staged pixel
  const int pp    = pix & 15;
  const int blane  = (klb & 1) ? (pp + 16) : pp;
  const int ldsOff = g * 512 + blane * 16 + ((klb >> 1) * 8);

  v8f z = {0.f,0.f,0.f,0.f,0.f,0.f,0.f,0.f};
  v8f acc00 = z, acc01 = z, acc02 = z, acc03 = z;
  v8f acc10 = z, acc11 = z, acc12 = z, acc13 = z;

  const size_t fbase   = (size_t)n * CIN * HW + (size_t)klb * 8 * HW;
  const size_t cstride = (size_t)32 * HW;   // channel chunk stride

  float r0, r1, r2, r3, r4, r5, r6, r7;

  for (int tap = 0; tap < 9; ++tap) {
    int t3  = tap / 3;
    int dy  = t3 - 1;
    int dx  = (tap - 3 * t3) - 1;
    int yy  = y + dy, xx = x + dx;
    bool ok = pval && yy >= 0 && yy < H && xx >= 0 && xx < W;
    int yyc = min(max(yy, 0), H - 1);
    int xxc = min(max(xx, 0), W - 1);
    const float* q = feat + fbase + (size_t)yyc * W + xxc;

    // preload chunk 0 of this tap
    r0 = q[0];              r1 = q[(size_t)1 * HW];
    r2 = q[(size_t)2 * HW]; r3 = q[(size_t)3 * HW];
    r4 = q[(size_t)4 * HW]; r5 = q[(size_t)5 * HW];
    r6 = q[(size_t)6 * HW]; r7 = q[(size_t)7 * HW];
    q += cstride;

#pragma unroll
    for (int cc = 0; cc < 8; ++cc) {
      const int ksg = tap * 8 + cc;
      // ---- pack the preloaded chunk, one b128 LDS store ----
      {
        float v0=r0,v1=r1,v2=r2,v3=r3,v4=r4,v5=r5,v6=r6,v7=r7;
        if (!ok) { v0=v1=v2=v3=v4=v5=v6=v7=0.f; }
        uint4 pk;
        pk.x = pk2bf(v0, v1);
        pk.y = pk2bf(v2, v3);
        pk.z = pk2bf(v4, v5);
        pk.w = pk2bf(v6, v7);
        *(uint4*)&ldsB[(ksg & 1) * 2048 + ldsOff] = pk;
      }
      __syncthreads();

      // ---- preload next chunk (overlaps with the WMMAs below) ----
      if (cc < 7) {
        r0 = q[0];              r1 = q[(size_t)1 * HW];
        r2 = q[(size_t)2 * HW]; r3 = q[(size_t)3 * HW];
        r4 = q[(size_t)4 * HW]; r5 = q[(size_t)5 * HW];
        r6 = q[(size_t)6 * HW]; r7 = q[(size_t)7 * HW];
        q += cstride;
      }

      const unsigned short* rb = &ldsB[(ksg & 1) * 2048 + lane * 16];
      const unsigned short* wbase =
          wA + ((size_t)(ksg * 16 + wave * 2) * 32 + lane) * 16;
      if (ksg + 1 < KSTEPS)
        __builtin_prefetch(wbase + 16 * 32 * 16, 0, 1);  // global_prefetch_b8

      v16bf b0 = *(const v16bf*)(rb);
      v16bf b1 = *(const v16bf*)(rb + 512);
      v16bf b2 = *(const v16bf*)(rb + 1024);
      v16bf b3 = *(const v16bf*)(rb + 1536);
      v16bf a0 = *(const v16bf*)(wbase);
      v16bf a1 = *(const v16bf*)(wbase + 512);

      acc00 = __builtin_amdgcn_wmma_f32_16x16x32_bf16(false, a0, false, b0,
                                                      (short)0, acc00, false, false);
      acc01 = __builtin_amdgcn_wmma_f32_16x16x32_bf16(false, a0, false, b1,
                                                      (short)0, acc01, false, false);
      acc02 = __builtin_amdgcn_wmma_f32_16x16x32_bf16(false, a0, false, b2,
                                                      (short)0, acc02, false, false);
      acc03 = __builtin_amdgcn_wmma_f32_16x16x32_bf16(false, a0, false, b3,
                                                      (short)0, acc03, false, false);
      acc10 = __builtin_amdgcn_wmma_f32_16x16x32_bf16(false, a1, false, b0,
                                                      (short)0, acc10, false, false);
      acc11 = __builtin_amdgcn_wmma_f32_16x16x32_bf16(false, a1, false, b1,
                                                      (short)0, acc11, false, false);
      acc12 = __builtin_amdgcn_wmma_f32_16x16x32_bf16(false, a1, false, b2,
                                                      (short)0, acc12, false, false);
      acc13 = __builtin_amdgcn_wmma_f32_16x16x32_bf16(false, a1, false, b3,
                                                      (short)0, acc13, false, false);
      // no trailing barrier: next chunk writes the other ldsB buffer
    }
  }
  __syncthreads();

  // ---- bias + ReLU, scatter t into B-fragment layout: frag (ks2, g) ----
  {
    int cp = lane & 15;                  // column (pixel) within tile
    int mh = (lane >> 4) * 8;            // lanes 16..31 hold M=8..15
#pragma unroll
    for (int v = 0; v < 8; ++v) {
      int ocA = wave * 32 + mh + v;        // M-tile 2w
      int ocB = wave * 32 + 16 + mh + v;   // M-tile 2w+1
      float bA = conv_b[ocA];
      float bB = conv_b[ocB];
      int ksA = ocA >> 5, klA = ocA & 31;
      int ksB = ocB >> 5, klB = ocB & 31;
      int blA = (klA < 16) ? cp : cp + 16;
      int blB = (klB < 16) ? cp : cp + 16;
      float t;
      t = acc00[v] + bA; ldsT[(ksA*4+0)*512 + blA*16 + (klA&15)] = f2bf(t>0.f?t:0.f);
      t = acc01[v] + bA; ldsT[(ksA*4+1)*512 + blA*16 + (klA&15)] = f2bf(t>0.f?t:0.f);
      t = acc02[v] + bA; ldsT[(ksA*4+2)*512 + blA*16 + (klA&15)] = f2bf(t>0.f?t:0.f);
      t = acc03[v] + bA; ldsT[(ksA*4+3)*512 + blA*16 + (klA&15)] = f2bf(t>0.f?t:0.f);
      t = acc10[v] + bB; ldsT[(ksB*4+0)*512 + blB*16 + (klB&15)] = f2bf(t>0.f?t:0.f);
      t = acc11[v] + bB; ldsT[(ksB*4+1)*512 + blB*16 + (klB&15)] = f2bf(t>0.f?t:0.f);
      t = acc12[v] + bB; ldsT[(ksB*4+2)*512 + blB*16 + (klB&15)] = f2bf(t>0.f?t:0.f);
      t = acc13[v] + bB; ldsT[(ksB*4+3)*512 + blB*16 + (klB&15)] = f2bf(t>0.f?t:0.f);
    }
  }
  __syncthreads();

  // ---- 1x1 heads as WMMA: waves 0..3 compute one 16(out)x16(pix) tile ----
  if (wave < 4) {
    const int gh = wave;                 // N-tile handled by this wave
    v8f hacc = z;
#pragma unroll
    for (int ks2 = 0; ks2 < 8; ++ks2) {
      v16bf a = *(const v16bf*)(headA + ks2 * 512 + lane * 16);
      v16bf b = *(const v16bf*)&ldsT[(ks2 * 4 + gh) * 512 + lane * 16];
      hacc = __builtin_amdgcn_wmma_f32_16x16x32_bf16(false, a, false, b,
                                                     (short)0, hacc, false, false);
    }
    int cp = lane & 15;
    int mh = (lane >> 4) * 8;
    int px = gh * 16 + cp;
    int pq = P0 + px;
    if (pq < total) {
      int nn = pq / HW;
      int hh = pq - nn * HW;
      size_t rowb = (size_t)nn * HW * A_ANCH + (size_t)hh * A_ANCH;
#pragma unroll
      for (int v = 0; v < 8; ++v) {
        int o = mh + v;
        if (o < 3) {
          sc[rowb + o] = hacc[v] + obj_b[o];
        } else if (o < 15) {
          int ch = o - 3;
          int a  = ch >> 2;
          int j  = ch & 3;
          dl[(rowb + a) * 4 + j] = hacc[v] + delta_b[ch];
        }
      }
    }
  }
}

// ---------------------------------------------------------------------------
// Top-k via 4-round MSD radix select (8-bit digits on sortable keys).
// state per n: [0]=prefix, [1]=rank, [2]=nGreater, [3]=c0, [4]=c1
// ---------------------------------------------------------------------------
__global__ void topk_init_kernel(int* __restrict__ hist,
                                 unsigned int* __restrict__ state, int k) {
  int n = blockIdx.x;
  hist[n * 256 + threadIdx.x] = 0;
  if (threadIdx.x == 0) {
    state[n * 8 + 0] = 0u;
    state[n * 8 + 1] = (unsigned int)k;
    state[n * 8 + 2] = 0u;
    state[n * 8 + 3] = 0u;
    state[n * 8 + 4] = 0u;
  }
}

__global__ void topk_hist_kernel(const float* __restrict__ scAll, int HWA,
                                 const unsigned int* __restrict__ state,
                                 int* __restrict__ hist, int round) {
  int n = blockIdx.y;
  unsigned int prefix = state[n * 8 + 0];
  __shared__ int lh[256];
  lh[threadIdx.x] = 0;
  __syncthreads();
  int i = blockIdx.x * 256 + threadIdx.x;
  if (i < HWA) {
    unsigned int u = fkey(scAll[(size_t)n * HWA + i]);
    bool part = (round == 0) || ((u >> (32 - 8 * round)) == prefix);
    if (part) atomicAdd(&lh[(u >> (24 - 8 * round)) & 255u], 1);
  }
  __syncthreads();
  if (lh[threadIdx.x]) atomicAdd(&hist[n * 256 + threadIdx.x], lh[threadIdx.x]);
}

__global__ void topk_scan_kernel(unsigned int* __restrict__ state,
                                 int* __restrict__ hist) {
  int n = blockIdx.x;
  int d = threadIdx.x;
  __shared__ int h[256];
  h[d] = hist[n * 256 + d];
  hist[n * 256 + d] = 0;                      // reset for next round
  int rank = (int)state[n * 8 + 1];
  unsigned int prefix = state[n * 8 + 0];
  __syncthreads();
  int gt = 0;
  for (int e = d + 1; e < 256; ++e) gt += h[e];
  if (gt < rank && rank <= gt + h[d]) {       // exactly one thread wins
    state[n * 8 + 0] = (prefix << 8) | (unsigned int)d;
    state[n * 8 + 1] = (unsigned int)(rank - gt);
    state[n * 8 + 2] += (unsigned int)gt;
  }
}

__global__ void topk_select_kernel(const float* __restrict__ scAll, int HWA,
                                   unsigned int* __restrict__ state,
                                   int* __restrict__ outIdx, int k) {
  int n = blockIdx.y;
  int i = blockIdx.x * 256 + threadIdx.x;
  if (i >= HWA) return;
  unsigned int T = state[n * 8 + 0];
  int nG = (int)state[n * 8 + 2];
  unsigned int u = fkey(scAll[(size_t)n * HWA + i]);
  if (u > T) {
    int pos = atomicAdd((int*)&state[n * 8 + 3], 1);
    outIdx[n * k + pos] = i;
  } else if (u == T) {
    int pos = nG + atomicAdd((int*)&state[n * 8 + 4], 1);
    if (pos < k) outIdx[n * k + pos] = i;
  }
}

// ---------------------------------------------------------------------------
// Decode: anchors (analytic), apply_deltas, clip, validity mask
// ---------------------------------------------------------------------------
__global__ void gather_boxes_kernel(const float* __restrict__ scAll,
                                    const float* __restrict__ dlAll,
                                    const int* __restrict__ idxbuf,
                                    int HWA, int k, int W, int stride,
                                    float asize, int Moff,
                                    float* __restrict__ boxesC,
                                    float* __restrict__ scoresC) {
  int t = blockIdx.x * blockDim.x + threadIdx.x;
  if (t >= NBATCH * k) return;
  int n = t / k;
  int j = t - n * k;
  int idx = idxbuf[n * k + j];
  float s = scAll[(size_t)n * HWA + idx];
  int a  = idx % A_ANCH;
  int hw = idx / A_ANCH;
  int x  = hw % W;
  int y  = hw / W;
  float ratio = (a == 0) ? 0.5f : ((a == 1) ? 1.0f : 2.0f);
  float wa = sqrtf(asize * asize / ratio);
  float ha = wa * ratio;
  float ctrx = (float)(x * stride);
  float ctry = (float)(y * stride);
  const float* d = &dlAll[((size_t)n * HWA + idx) * 4];
  float dx = d[0], dy = d[1];
  float dw = fminf(d[2], SCALE_CLAMP);
  float dh = fminf(d[3], SCALE_CLAMP);
  float pcx = dx * wa + ctrx;
  float pcy = dy * ha + ctry;
  float pw  = expf(dw) * wa;
  float ph  = expf(dh) * ha;
  float x1 = fminf(fmaxf(pcx - 0.5f * pw, 0.f), IMG_W);
  float y1 = fminf(fmaxf(pcy - 0.5f * ph, 0.f), IMG_H);
  float x2 = fminf(fmaxf(pcx + 0.5f * pw, 0.f), IMG_W);
  float y2 = fminf(fmaxf(pcy + 0.5f * ph, 0.f), IMG_H);
  bool valid = ((x2 - x1) > 0.f) && ((y2 - y1) > 0.f);
  int m = Moff + j;
  float* bo = &boxesC[((size_t)n * MTOT + m) * 4];
  bo[0] = x1; bo[1] = y1; bo[2] = x2; bo[3] = y2;
  scoresC[(size_t)n * MTOT + m] = valid ? s : NEG_INF;
}

// ---------------------------------------------------------------------------
// Greedy fixed-iteration NMS: 1000 argmax+suppress rounds, one block per n.
// Tie-break matches jnp.argmax (lowest index).
// ---------------------------------------------------------------------------
__global__ __launch_bounds__(256)
void nms_kernel(const float* __restrict__ boxesC,
                const float* __restrict__ scoresC,
                float* __restrict__ out) {
  __shared__ float ss[MTOT];
  __shared__ float rv[256];
  __shared__ int   ri[256];
  int n = blockIdx.x;
  int tid = threadIdx.x;
  for (int i = tid; i < MTOT; i += 256) ss[i] = scoresC[(size_t)n * MTOT + i];
  __syncthreads();
  const float* bb = boxesC + (size_t)n * MTOT * 4;
  for (int it = 0; it < POSTK; ++it) {
    float bv = -3e38f;
    int   bi = 0;
    for (int i = tid; i < MTOT; i += 256) {
      float v = ss[i];
      if (v > bv) { bv = v; bi = i; }
    }
    rv[tid] = bv; ri[tid] = bi;
    __syncthreads();
    for (int sft = 128; sft > 0; sft >>= 1) {
      if (tid < sft) {
        if (rv[tid + sft] > rv[tid] ||
            (rv[tid + sft] == rv[tid] && ri[tid + sft] < ri[tid])) {
          rv[tid] = rv[tid + sft];
          ri[tid] = ri[tid + sft];
        }
      }
      __syncthreads();
    }
    int   sel = ri[0];
    float sv  = rv[0];
    float bx0 = bb[sel * 4 + 0], by0 = bb[sel * 4 + 1];
    float bx1 = bb[sel * 4 + 2], by1 = bb[sel * 4 + 3];
    bool valid = sv > (NEG_INF * 0.5f);
    if (tid == 0) {
      float* o = out + ((size_t)n * POSTK + it) * 5;
      o[0] = valid ? bx0 : 0.f;
      o[1] = valid ? by0 : 0.f;
      o[2] = valid ? bx1 : 0.f;
      o[3] = valid ? by1 : 0.f;
      o[4] = valid ? sv  : NEG_INF;
    }
    __syncthreads();
    float area1 = (bx1 - bx0) * (by1 - by0);
    for (int i = tid; i < MTOT; i += 256) {
      float ax0 = bb[i * 4 + 0], ay0 = bb[i * 4 + 1];
      float ax1 = bb[i * 4 + 2], ay1 = bb[i * 4 + 3];
      float lx = fmaxf(bx0, ax0), ly = fmaxf(by0, ay0);
      float rx = fminf(bx1, ax1), ry = fminf(by1, ay1);
      float iw = fmaxf(rx - lx, 0.f), ih = fmaxf(ry - ly, 0.f);
      float inter = iw * ih;
      float a2 = (ax1 - ax0) * (ay1 - ay0);
      float iou = inter / fmaxf(area1 + a2 - inter, 1e-9f);
      if (iou > 0.7f) ss[i] = NEG_INF;
    }
    __syncthreads();
  }
}

// ---------------------------------------------------------------------------
// Host launcher
// ---------------------------------------------------------------------------
extern "C" void kernel_launch(void* const* d_in, const int* in_sizes, int n_in,
                              void* d_out, int out_size, void* d_ws,
                              size_t ws_size, hipStream_t stream) {
  (void)in_sizes; (void)n_in; (void)out_size; (void)ws_size;

  static const int Hs[NLEV]      = {200, 100, 50, 25, 13};
  static const int WsA[NLEV]     = {304, 152, 76, 38, 19};
  static const int HWs[NLEV]     = {60800, 15200, 3800, 950, 247};
  static const int strides[NLEV] = {4, 8, 16, 32, 64};
  static const float sizesA[NLEV]= {32.f, 64.f, 128.f, 256.f, 512.f};
  static const int Kk[NLEV]      = {1000, 1000, 1000, 1000, 741};
  static const int Moffs[NLEV]   = {0, 1000, 2000, 3000, 4000};

  const float* feats[NLEV] = {
      (const float*)d_in[0], (const float*)d_in[1], (const float*)d_in[2],
      (const float*)d_in[3], (const float*)d_in[4]};
  const float* conv_w  = (const float*)d_in[5];
  const float* conv_b  = (const float*)d_in[6];
  const float* obj_w   = (const float*)d_in[7];
  const float* obj_b   = (const float*)d_in[8];
  const float* delta_w = (const float*)d_in[9];
  const float* delta_b = (const float*)d_in[10];
  float* out = (float*)d_out;

  // ---- workspace carve-up (bytes, 256B aligned) ----
  char* wsb = (char*)d_ws;
  size_t off = 0;
  auto alloc = [&](size_t bytes) {
    size_t cur = off;
    off += (bytes + 255) & ~(size_t)255;
    return cur;
  };
  size_t off_wA = alloc((size_t)KSTEPS * 16 * 512 * sizeof(unsigned short));
  size_t off_hA = alloc((size_t)8 * 512 * sizeof(unsigned short));
  size_t off_sc[NLEV], off_dl[NLEV], off_idx[NLEV];
  for (int l = 0; l < NLEV; ++l)
    off_sc[l] = alloc((size_t)NBATCH * HWs[l] * A_ANCH * sizeof(float));
  for (int l = 0; l < NLEV; ++l)
    off_dl[l] = alloc((size_t)NBATCH * HWs[l] * A_ANCH * 4 * sizeof(float));
  size_t off_hist  = alloc((size_t)NLEV * NBATCH * 256 * sizeof(int));
  size_t off_state = alloc((size_t)NLEV * NBATCH * 8 * sizeof(unsigned int));
  for (int l = 0; l < NLEV; ++l)
    off_idx[l] = alloc((size_t)NBATCH * Kk[l] * sizeof(int));
  size_t off_boxc = alloc((size_t)NBATCH * MTOT * 4 * sizeof(float));
  size_t off_scc  = alloc((size_t)NBATCH * MTOT * sizeof(float));

  unsigned short* wA = (unsigned short*)(wsb + off_wA);
  unsigned short* hA = (unsigned short*)(wsb + off_hA);

  // 1) pack conv + head weights into WMMA A-fragments
  {
    int tot = KSTEPS * 16 * 32 * 16;
    pack_wA_kernel<<<(tot + 255) / 256, 256, 0, stream>>>(conv_w, wA);
    pack_headA_kernel<<<16, 256, 0, stream>>>(obj_w, delta_w, hA);
  }

  // 2) fused conv + heads per level
  for (int l = 0; l < NLEV; ++l) {
    int tiles = (NBATCH * HWs[l] + NPIX - 1) / NPIX;
    conv_rpn_wmma_kernel<<<tiles, NTHR, 0, stream>>>(
        feats[l], wA, hA, conv_b, obj_b, delta_b,
        (float*)(wsb + off_sc[l]), (float*)(wsb + off_dl[l]),
        Hs[l], WsA[l]);
  }

  // 3) per-level top-k radix select + box decode
  for (int l = 0; l < NLEV; ++l) {
    int HWA = HWs[l] * A_ANCH;
    int k   = Kk[l];
    int* hist           = (int*)(wsb + off_hist) + l * NBATCH * 256;
    unsigned int* state = (unsigned int*)(wsb + off_state) + l * NBATCH * 8;
    const float* scl    = (const float*)(wsb + off_sc[l]);
    const float* dll    = (const float*)(wsb + off_dl[l]);
    int* idxb           = (int*)(wsb + off_idx[l]);

    topk_init_kernel<<<NBATCH, 256, 0, stream>>>(hist, state, k);
    dim3 g((HWA + 255) / 256, NBATCH);
    for (int r = 0; r < 4; ++r) {
      topk_hist_kernel<<<g, 256, 0, stream>>>(scl, HWA, state, hist, r);
      topk_scan_kernel<<<NBATCH, 256, 0, stream>>>(state, hist);
    }
    topk_select_kernel<<<g, 256, 0, stream>>>(scl, HWA, state, idxb, k);
    gather_boxes_kernel<<<(NBATCH * k + 255) / 256, 256, 0, stream>>>(
        scl, dll, idxb, HWA, k, WsA[l], strides[l], sizesA[l], Moffs[l],
        (float*)(wsb + off_boxc), (float*)(wsb + off_scc));
  }

  // 4) greedy NMS, one block per batch element
  nms_kernel<<<NBATCH, 256, 0, stream>>>((const float*)(wsb + off_boxc),
                                         (const float*)(wsb + off_scc), out);
}
